// GPTJAttention_32315333935800
// MI455X (gfx1250) — matrix operand order repaired and checked by reading, not compiled, not run on hardware
//
#include <hip/hip_runtime.h>
#include <stdint.h>

#define B_ 2
#define S_ 2048
#define D_ 4096
#define H_ 16
#define HD_ 256
#define ROT_ 64

typedef __attribute__((ext_vector_type(16))) __bf16 v16bf;
typedef __attribute__((ext_vector_type(8)))  float  v8f;
typedef __attribute__((ext_vector_type(4)))  unsigned int u32x4;
typedef __attribute__((ext_vector_type(8)))  int i32x8;
typedef __attribute__((ext_vector_type(4)))  int i32x4;

union FragBF { v16bf v; uint4 u[2]; unsigned short h[16]; };
union FragF  { v8f  v; float f[8]; };

__device__ __forceinline__ unsigned short f2bf(float f) {
  union { float f; unsigned int u; } a; a.f = f;
  unsigned int u = a.u;
  u += 0x7fffu + ((u >> 16) & 1u);   // round-to-nearest-even
  return (unsigned short)(u >> 16);
}

// --------------------------------------------------------------------------
// Tensor Data Mover availability (TDM: tensor_load_to_lds + s_wait_tensorcnt)
// --------------------------------------------------------------------------
#if defined(__has_builtin)
#  if __has_builtin(__builtin_amdgcn_tensor_load_to_lds) && \
      __has_builtin(__builtin_amdgcn_s_wait_tensorcnt)
#    define HAVE_TDM 1
#  endif
#endif
#ifndef HAVE_TDM
#  define HAVE_TDM 0
#endif

#if HAVE_TDM
// Issue one 2D TDM load: tile_d1 rows x tile_d0 elements (bf16, 2 bytes each),
// row stride stride0 (elements) in global memory, packed contiguously in LDS.
// D# layout per CDNA5 ISA ch.8 (group0 128b, group1 256b; groups 2/3 zero).
__device__ __forceinline__ void tdm_load_2d(unsigned lds_off, const void* gptr,
                                            unsigned tile_d0, unsigned tile_d1,
                                            unsigned long long stride0) {
  unsigned long long ga = (unsigned long long)gptr;
  u32x4 g0;
  g0.x = 0x1u;                                   // count=1, user descriptor
  g0.y = lds_off;                                // lds_addr (bytes)
  g0.z = (unsigned)(ga & 0xFFFFFFFFu);           // global_addr[31:0]
  g0.w = (unsigned)((ga >> 32) & 0x01FFFFFFu)    // global_addr[56:32]
         | (2u << 30);                           // type=2 ("image")
  unsigned td0 = (unsigned)stride0;              // tensor_dim0 (elements)
  unsigned td1 = 1u << 20;                       // tensor_dim1 (generous)
  i32x8 g1;
  g1[0] = (int)(1u << 16);                       // wg_mask=0, data_size=1 (2B)
  g1[1] = (int)((td0 & 0xFFFFu) << 16);          // tensor_dim0 lo16 @ bits[63:48]
  g1[2] = (int)((td0 >> 16) | ((td1 & 0xFFFFu) << 16));   // td0 hi16 | td1 lo16
  g1[3] = (int)((td1 >> 16) | (tile_d0 << 16));  // td1 hi16 | tile_dim0
  g1[4] = (int)(tile_d1 & 0xFFFFu);              // tile_dim1 | tile_dim2=0
  g1[5] = (int)(stride0 & 0xFFFFFFFFull);        // tensor_dim0_stride[31:0]
  g1[6] = (int)((stride0 >> 32) & 0xFFFFull);    // stride0[47:32] | stride1 lo=0
  g1[7] = 0;
  i32x4 z4 = {0, 0, 0, 0};
#if __clang_major__ >= 23
  i32x8 z8 = {0, 0, 0, 0, 0, 0, 0, 0};
  __builtin_amdgcn_tensor_load_to_lds(g0, g1, z4, z4, z8, 0);
#else
  __builtin_amdgcn_tensor_load_to_lds(g0, g1, z4, z4, 0);
#endif
}
#endif  // HAVE_TDM

// ---------------------------------------------------------------------------
// float -> bf16 cast
// ---------------------------------------------------------------------------
__global__ void cast_bf16_kernel(const float* __restrict__ src,
                                 unsigned short* __restrict__ dst, int n) {
  int i = blockIdx.x * blockDim.x + threadIdx.x;
  int stride = gridDim.x * blockDim.x;
  for (; i < n; i += stride) dst[i] = f2bf(src[i]);
}

// ---------------------------------------------------------------------------
// C[m,n] = sum_k A[m,k] * W[n,k]   (A: bf16 MxK row-major, W: bf16 NxK row-major)
// 128x128 block tile, K-step 32, 8 waves arranged 4(M) x 2(N).
// Tiles staged into double-buffered LDS by the Tensor Data Mover (wave 0
// issues, TENSORcnt + workgroup barrier publishes); falls back to manual
// VMEM staging when the TDM builtin is unavailable.
// ---------------------------------------------------------------------------
__global__ __launch_bounds__(256) void gemm_bf16_nt(
    const unsigned short* __restrict__ A, const unsigned short* __restrict__ W,
    float* __restrict__ C, int M, int N, int K) {
  __shared__ __align__(16) unsigned short As[2][128 * 32];
  __shared__ __align__(16) unsigned short Bs[2][128 * 32];

  int tid = threadIdx.x;
  int lane = tid & 31, wid = tid >> 5;
  int wm = wid & 3, wn = wid >> 2;
  int laneHalf = lane >> 4, ln = lane & 15;
  int m0 = blockIdx.x * 128;
  int n0 = blockIdx.y * 128;

  FragF acc[2][4];
  for (int i = 0; i < 2; i++)
    for (int j = 0; j < 4; j++)
      for (int r = 0; r < 8; r++) acc[i][j].f[r] = 0.0f;

  const int ksteps = K >> 5;

#if HAVE_TDM
  unsigned ldsA0 = (unsigned)(uintptr_t)(void*)&As[0][0];
  unsigned ldsA1 = (unsigned)(uintptr_t)(void*)&As[1][0];
  unsigned ldsB0 = (unsigned)(uintptr_t)(void*)&Bs[0][0];
  unsigned ldsB1 = (unsigned)(uintptr_t)(void*)&Bs[1][0];
  if (wid == 0) {
    tdm_load_2d(ldsA0, A + (size_t)m0 * K, 32, 128, (unsigned long long)K);
    tdm_load_2d(ldsB0, W + (size_t)n0 * K, 32, 128, (unsigned long long)K);
  }
#else
  int lrow = tid >> 1;          // 0..127
  int lseg = (tid & 1) * 16;    // 0 or 16 (halves)
#endif

  for (int kt = 0; kt < ksteps; kt++) {
    int cur = kt & 1;
#if HAVE_TDM
    if (wid == 0) {
      if (kt + 1 < ksteps) {
        int nxt = cur ^ 1;
        size_t koff = (size_t)(kt + 1) << 5;
        tdm_load_2d(nxt ? ldsA1 : ldsA0, A + (size_t)m0 * K + koff, 32, 128,
                    (unsigned long long)K);
        tdm_load_2d(nxt ? ldsB1 : ldsB0, W + (size_t)n0 * K + koff, 32, 128,
                    (unsigned long long)K);
        __builtin_amdgcn_s_wait_tensorcnt(2);   // current pair retired
      } else {
        __builtin_amdgcn_s_wait_tensorcnt(0);
      }
    }
    __syncthreads();                            // publish tile to all waves
#else
    int k0 = kt << 5;
    const uint4* ag = (const uint4*)(A + (size_t)(m0 + lrow) * K + k0 + lseg);
    const uint4* wg = (const uint4*)(W + (size_t)(n0 + lrow) * K + k0 + lseg);
    uint4 a0 = ag[0], a1 = ag[1];
    uint4 w0 = wg[0], w1 = wg[1];
    __syncthreads();
    *(uint4*)&As[cur][lrow * 32 + lseg]     = a0;
    *(uint4*)&As[cur][lrow * 32 + lseg + 8] = a1;
    *(uint4*)&Bs[cur][lrow * 32 + lseg]     = w0;
    *(uint4*)&Bs[cur][lrow * 32 + lseg + 8] = w1;
    __syncthreads();
#endif

    FragBF afrag[2], bfrag[4];
    for (int i = 0; i < 2; i++) {
      int row = wm * 32 + i * 16 + ln;
      int koff = laneHalf ? 8 : 0;         // A layout: lanes0-15 K{0..7,16..23}
      afrag[i].u[0] = *(const uint4*)&As[cur][row * 32 + koff];
      afrag[i].u[1] = *(const uint4*)&As[cur][row * 32 + koff + 16];
    }
    for (int j = 0; j < 4; j++) {
      int nrow = wn * 64 + j * 16 + ln;
      int koff = laneHalf ? 8 : 0;         // B layout: lanes0-15 K{0..7,16..23}
      bfrag[j].u[0] = *(const uint4*)&Bs[cur][nrow * 32 + koff];
      bfrag[j].u[1] = *(const uint4*)&Bs[cur][nrow * 32 + koff + 16];
    }
    for (int i = 0; i < 2; i++)
      for (int j = 0; j < 4; j++)
        acc[i][j].v = __builtin_amdgcn_wmma_f32_16x16x32_bf16(
            false, afrag[i].v, false, bfrag[j].v, (short)0, acc[i][j].v, false, false);

#if HAVE_TDM
    __syncthreads();          // all reads of buffer `cur` done before reissue
#endif
  }

  for (int i = 0; i < 2; i++) {
    int mrow0 = m0 + wm * 32 + i * 16 + laneHalf * 8;
    for (int j = 0; j < 4; j++) {
      int col = n0 + wn * 64 + j * 16 + ln;
      for (int r = 0; r < 8; r++)
        C[(size_t)(mrow0 + r) * N + col] = acc[i][j].f[r];
    }
  }
}

// ---------------------------------------------------------------------------
// RoPE (GPT-J rotate_every_two) + transpose to [b,h,s,hd] + bf16 pack
// ---------------------------------------------------------------------------
__global__ void rope_pack_kernel(const float* __restrict__ Qf, const float* __restrict__ Kf,
                                 const float* __restrict__ Vf, const float* __restrict__ sinp,
                                 const float* __restrict__ cosp,
                                 unsigned short* __restrict__ Qb,
                                 unsigned short* __restrict__ Kb,
                                 unsigned short* __restrict__ Vb) {
  long tid = blockIdx.x * (long)blockDim.x + threadIdx.x;
  long total = (long)B_ * S_ * H_ * (HD_ / 2);
  if (tid >= total) return;
  int pair = (int)(tid % (HD_ / 2));
  long t = tid / (HD_ / 2);
  int h = (int)(t % H_); t /= H_;
  int s = (int)(t % S_); t /= S_;
  int b = (int)t;
  int d0 = pair * 2;
  size_t src = ((size_t)(b * S_ + s)) * D_ + h * HD_ + d0;
  size_t dst = ((size_t)((b * H_ + h) * S_ + s)) * HD_ + d0;
  float q0 = Qf[src], q1 = Qf[src + 1];
  float k0 = Kf[src], k1 = Kf[src + 1];
  float v0 = Vf[src], v1 = Vf[src + 1];
  if (d0 < ROT_) {
    float c  = cosp[s * ROT_ + d0];
    float sn = sinp[s * ROT_ + d0];
    float q0r = q0 * c - q1 * sn, q1r = q1 * c + q0 * sn;
    float k0r = k0 * c - k1 * sn, k1r = k1 * c + k0 * sn;
    q0 = q0r; q1 = q1r; k0 = k0r; k1 = k1r;
  }
  Qb[dst] = f2bf(q0); Qb[dst + 1] = f2bf(q1);
  Kb[dst] = f2bf(k0); Kb[dst + 1] = f2bf(k1);
  Vb[dst] = f2bf(v0); Vb[dst + 1] = f2bf(v1);
}

// ---------------------------------------------------------------------------
// Flash attention. Block = (b*H+h, 128 queries), 8 waves x 16 query rows.
// KV tiles of 32 keys staged in LDS. Online softmax, WMMA for QK^T and P@V.
// Output context as bf16 in [b, s, h*HD+d] layout (transpose for free).
// ---------------------------------------------------------------------------
__global__ __launch_bounds__(256) void attn_kernel(
    const unsigned short* __restrict__ Qb, const unsigned short* __restrict__ Kb,
    const unsigned short* __restrict__ Vb, unsigned short* __restrict__ Ctx) {
  __shared__ __align__(16) unsigned short Ks[32 * 256];   // [kk][d]  16KB
  __shared__ __align__(16) unsigned short Vt[256 * 32];   // [d][kk]  16KB
  __shared__ __align__(16) unsigned short Ps[8][16 * 32]; // per-wave P  8KB

  int tid = threadIdx.x;
  int lane = tid & 31, wid = tid >> 5;
  int laneHalf = lane >> 4, ln = lane & 15;
  int qb = blockIdx.x * 128;
  int bh = blockIdx.y;
  const unsigned short* Qbh = Qb + (size_t)bh * S_ * HD_;
  const unsigned short* Kbh = Kb + (size_t)bh * S_ * HD_;
  const unsigned short* Vbh = Vb + (size_t)bh * S_ * HD_;

  FragBF qfrag[8];
  {
    int row = qb + wid * 16 + ln;
    const unsigned short* qr = Qbh + (size_t)row * HD_;
    int koff = laneHalf ? 8 : 0;
    for (int c = 0; c < 8; c++) {
      qfrag[c].u[0] = *(const uint4*)&qr[c * 32 + koff];
      qfrag[c].u[1] = *(const uint4*)&qr[c * 32 + koff + 16];
    }
  }

  FragF o[16];
  for (int t = 0; t < 16; t++)
    for (int r = 0; r < 8; r++) o[t].f[r] = 0.0f;
  float m_run[8], l_run[8];
  for (int r = 0; r < 8; r++) { m_run[r] = -1e30f; l_run[r] = 0.0f; }
  const float inv_scale = 0.0625f;   // 1/sqrt(256)

  int kv_end = qb + 128;
  for (int kv0 = 0; kv0 < kv_end; kv0 += 32) {
    for (int i = 0; i < 4; i++) {
      int idx = tid + 256 * i;
      int hidx = idx * 8;
      int row = hidx >> 8;
      int col = hidx & 255;
      uint4 kd = *(const uint4*)&Kbh[(size_t)(kv0 + row) * HD_ + col];
      uint4 vd = *(const uint4*)&Vbh[(size_t)(kv0 + row) * HD_ + col];
      *(uint4*)&Ks[row * 256 + col] = kd;
      unsigned short vh[8];
      *(uint4*)vh = vd;
      for (int j = 0; j < 8; j++) Vt[(col + j) * 32 + row] = vh[j];
    }
    __syncthreads();

    FragF sfr[2];
    for (int j = 0; j < 2; j++)
      for (int r = 0; r < 8; r++) sfr[j].f[r] = 0.0f;
    for (int c = 0; c < 8; c++) {
      for (int j = 0; j < 2; j++) {
        FragBF bfr;
        int kkrow = j * 16 + ln;
        int koff = c * 32 + (laneHalf ? 8 : 0);
        bfr.u[0] = *(const uint4*)&Ks[kkrow * 256 + koff];
        bfr.u[1] = *(const uint4*)&Ks[kkrow * 256 + koff + 16];
        sfr[j].v = __builtin_amdgcn_wmma_f32_16x16x32_bf16(
            false, qfrag[c].v, false, bfr.v, (short)0, sfr[j].v, false, false);
      }
    }
    for (int j = 0; j < 2; j++) {
      int key = kv0 + j * 16 + ln;
      for (int r = 0; r < 8; r++) {
        int qrow = qb + wid * 16 + laneHalf * 8 + r;
        float v = sfr[j].f[r] * inv_scale;
        sfr[j].f[r] = (key <= qrow) ? v : -1e30f;
      }
    }

    float alpha[8];
    for (int r = 0; r < 8; r++) {
      float mx = fmaxf(sfr[0].f[r], sfr[1].f[r]);
      for (int d = 1; d < 16; d <<= 1) mx = fmaxf(mx, __shfl_xor(mx, d, 16));
      float m_new = fmaxf(m_run[r], mx);
      float a = __expf(m_run[r] - m_new);
      float rs = 0.0f;
      for (int j = 0; j < 2; j++) {
        float pv = __expf(sfr[j].f[r] - m_new);
        sfr[j].f[r] = pv;
        rs += pv;
      }
      for (int d = 1; d < 16; d <<= 1) rs += __shfl_xor(rs, d, 16);
      l_run[r] = l_run[r] * a + rs;
      m_run[r] = m_new;
      alpha[r] = a;
    }
    for (int t = 0; t < 16; t++)
      for (int r = 0; r < 8; r++) o[t].f[r] *= alpha[r];

    for (int j = 0; j < 2; j++)
      for (int r = 0; r < 8; r++)
        Ps[wid][(laneHalf * 8 + r) * 32 + j * 16 + ln] = f2bf(sfr[j].f[r]);
    __syncthreads();

    FragBF pf;
    {
      int koff = laneHalf ? 8 : 0;
      pf.u[0] = *(const uint4*)&Ps[wid][ln * 32 + koff];
      pf.u[1] = *(const uint4*)&Ps[wid][ln * 32 + koff + 16];
    }
    for (int t = 0; t < 16; t++) {
      FragBF vfr;
      int drow = t * 16 + ln;
      int koff = laneHalf ? 8 : 0;
      vfr.u[0] = *(const uint4*)&Vt[drow * 32 + koff];
      vfr.u[1] = *(const uint4*)&Vt[drow * 32 + koff + 16];
      o[t].v = __builtin_amdgcn_wmma_f32_16x16x32_bf16(
          false, pf.v, false, vfr.v, (short)0, o[t].v, false, false);
    }
    __syncthreads();
  }

  int b = bh >> 4, h = bh & 15;
  for (int r = 0; r < 8; r++) {
    float inv_l = 1.0f / l_run[r];
    int srow = qb + wid * 16 + laneHalf * 8 + r;
    size_t base = ((size_t)(b * S_ + srow)) * D_ + h * HD_;
    for (int t = 0; t < 16; t++)
      Ctx[base + t * 16 + ln] = f2bf(o[t].f[r] * inv_l);
  }
}

// ---------------------------------------------------------------------------
extern "C" void kernel_launch(void* const* d_in, const int* in_sizes, int n_in,
                              void* d_out, int out_size, void* d_ws, size_t ws_size,
                              hipStream_t stream) {
  const float* X    = (const float*)d_in[0];
  const float* sinp = (const float*)d_in[1];
  const float* cosp = (const float*)d_in[2];
  const float* Wq   = (const float*)d_in[3];
  const float* Wk   = (const float*)d_in[4];
  const float* Wv   = (const float*)d_in[5];
  const float* Wo   = (const float*)d_in[6];
  float* out = (float*)d_out;

  const size_t NX = (size_t)B_ * S_ * D_;   // 16,777,216
  const size_t NW = (size_t)D_ * D_;        // 16,777,216
  char* p = (char*)d_ws;
  unsigned short* Xb  = (unsigned short*)p; p += NX * 2;
  unsigned short* Wqb = (unsigned short*)p; p += NW * 2;
  unsigned short* Wkb = (unsigned short*)p; p += NW * 2;
  unsigned short* Wvb = (unsigned short*)p; p += NW * 2;
  unsigned short* Wob = (unsigned short*)p; p += NW * 2;
  float* Qf = (float*)p; p += NX * 4;
  float* Kf = (float*)p; p += NX * 4;
  float* Vf = (float*)p; p += NX * 4;
  unsigned short* Qbh = (unsigned short*)p; p += NX * 2;
  unsigned short* Kbh = (unsigned short*)p; p += NX * 2;
  unsigned short* Vbh = (unsigned short*)p; p += NX * 2;
  unsigned short* Ctx = (unsigned short*)p; p += NX * 2;

  const int M = B_ * S_;   // 4096
  const int N = D_, K = D_;

  int castBlocks = (int)((NX + 255) / 256);
  cast_bf16_kernel<<<castBlocks, 256, 0, stream>>>(X,  Xb,  (int)NX);
  cast_bf16_kernel<<<castBlocks, 256, 0, stream>>>(Wq, Wqb, (int)NW);
  cast_bf16_kernel<<<castBlocks, 256, 0, stream>>>(Wk, Wkb, (int)NW);
  cast_bf16_kernel<<<castBlocks, 256, 0, stream>>>(Wv, Wvb, (int)NW);
  cast_bf16_kernel<<<castBlocks, 256, 0, stream>>>(Wo, Wob, (int)NW);

  dim3 gg(M / 128, N / 128);
  gemm_bf16_nt<<<gg, 256, 0, stream>>>(Xb, Wqb, Qf, M, N, K);
  gemm_bf16_nt<<<gg, 256, 0, stream>>>(Xb, Wkb, Kf, M, N, K);
  gemm_bf16_nt<<<gg, 256, 0, stream>>>(Xb, Wvb, Vf, M, N, K);

  long pairs = (long)B_ * S_ * H_ * (HD_ / 2);
  rope_pack_kernel<<<(int)((pairs + 255) / 256), 256, 0, stream>>>(
      Qf, Kf, Vf, sinp, cosp, Qbh, Kbh, Vbh);

  dim3 ga(S_ / 128, B_ * H_);
  attn_kernel<<<ga, 256, 0, stream>>>(Qbh, Kbh, Vbh, Ctx);

  gemm_bf16_nt<<<gg, 256, 0, stream>>>(Ctx, Wob, out, M, N, K);
}
